// FEMBA_36447092473922
// MI455X (gfx1250) — matrix operand order repaired and checked by reading, not compiled, not run on hardware
//
#include <hip/hip_runtime.h>
#include <cstdint>
#include <cstddef>

// ---------------------------------------------------------------------------
// FEMBA-style bidirectional Mamba autoencoder for MI455X (gfx1250, wave32).
// Dense projections: v_wmma_f32_16x16x32_bf16 with async-to-LDS A staging
// (global_load_async_to_lds_b128 + s_wait_asynccnt, double buffered).
// ---------------------------------------------------------------------------

typedef __bf16 bf16;
typedef __attribute__((ext_vector_type(16))) __bf16 v16bf;
typedef __attribute__((ext_vector_type(8)))  float  v8f;

struct alignas(16) I4 { int x, y, z, w; };
union FragU { v16bf v; I4 q[2]; };

#define BATCH 8
#define LSEQ  256
#define TOK   2048          // BATCH * LSEQ
#define DMOD  2048
#define DINN  4096
#define DSTT  16
#define DTRR  128
#define XPN   160           // DTR + 2*DS
#define NPROJ 8192          // 2*DIN
#define AROW  40            // padded LDS row stride (elements) -> bank-conflict floor

__device__ __forceinline__ bf16 to_bf16(float f) {
  unsigned u = __builtin_bit_cast(unsigned, f);
  u += 0x7fffu + ((u >> 16) & 1u);                 // round-to-nearest-even
  unsigned short h = (unsigned short)(u >> 16);
  return __builtin_bit_cast(bf16, h);
}
__device__ __forceinline__ float silu_(float x) { return x / (1.f + __expf(-x)); }

__device__ __forceinline__ v8f wmma_bf16(const FragU& a, const FragU& b, v8f c) {
  return __builtin_amdgcn_wmma_f32_16x16x32_bf16(false, a.v, false, b.v,
                                                 (short)0, c, false, false);
}

// ---------------------------------------------------------------------------
// Generic converters
// ---------------------------------------------------------------------------
__global__ void cvt_bf16(const float* __restrict__ src, bf16* __restrict__ dst, size_t n) {
  size_t i = (size_t)blockIdx.x * blockDim.x + threadIdx.x;
  if (i < n) dst[i] = to_bf16(src[i]);
}
__global__ void cvt_bf16_strided(const float* __restrict__ src, bf16* __restrict__ dst,
                                 int rows, int cols, int src_stride) {
  size_t i = (size_t)blockIdx.x * blockDim.x + threadIdx.x;
  if (i >= (size_t)rows * cols) return;
  int r = (int)(i / cols), c = (int)(i % cols);
  dst[i] = to_bf16(src[(size_t)r * src_stride + c]);
}
__global__ void copy_f32(const float* __restrict__ src, float* __restrict__ dst, size_t n) {
  size_t i = (size_t)blockIdx.x * blockDim.x + threadIdx.x;
  if (i < n) dst[i] = src[i];
}

// ---------------------------------------------------------------------------
// Patch embedding: h[b,l,d] = sum_{2x16 patch} mask?0:x * proj_w + proj_b + pos
// ---------------------------------------------------------------------------
__global__ void patch_embed(const float* __restrict__ x, const unsigned char* __restrict__ mask,
                            const float* __restrict__ pw, const float* __restrict__ pb,
                            const float* __restrict__ pos, float* __restrict__ h) {
  size_t i = (size_t)blockIdx.x * blockDim.x + threadIdx.x;   // TOK * DMOD
  if (i >= (size_t)TOK * DMOD) return;
  int d = (int)(i & (DMOD - 1));
  size_t token = i >> 11;
  int l = (int)(token & (LSEQ - 1));
  int b = (int)(token >> 8);
  int e = d >> 5, g0 = d & 31;
  float acc = pb[e];
  #pragma unroll
  for (int ii = 0; ii < 2; ++ii) {
    size_t rowbase = ((size_t)(b * 64 + 2 * g0 + ii)) * 4096 + (size_t)l * 16;
    #pragma unroll
    for (int j = 0; j < 16; ++j) {
      size_t xi = rowbase + j;
      float xv = mask[xi] ? 0.f : x[xi];
      acc = fmaf(xv, pw[e * 32 + ii * 16 + j], acc);
    }
  }
  h[i] = acc + pos[(size_t)l * DMOD + d];
}

// ---------------------------------------------------------------------------
// bf16 WMMA GEMM:  Y[M,N] = X[M,K] * W[N,K]^T (+bias[n]) (act: 0=none,1=softplus)
// Block = 8 waves sharing one 64-row M stripe; the 64x32 A tile is staged into
// LDS (double buffered) with global_load_async_to_lds_b128. Each wave owns a
// 64(M) x 32(N) output tile: 8 WMMAs per 32-K step.
// ---------------------------------------------------------------------------
__global__ __launch_bounds__(256)
void gemm_bf16_wmma(const bf16* __restrict__ X, const bf16* __restrict__ W,
                    const float* __restrict__ bias, float* __restrict__ Y,
                    int M, int N, int K, int beta, int act) {
  __shared__ bf16 Abuf[2][64 * AROW];

  const int wid  = threadIdx.x >> 5;
  const int lane = threadIdx.x & 31;
  const int tilesN = N >> 5;                     // 32-wide N tiles
  const int tn = blockIdx.x * 8 + wid;
  const int mbase = blockIdx.y << 6;
  const bool active = (tn < tilesN);
  const int nbase = tn << 5;
  const int idx = lane & 15, grp = lane >> 4;

  // cooperative A staging: thread t loads 16 B: row = t/4, chunk = t%4
  const int srow = threadIdx.x >> 2, schunk = threadIdx.x & 3;
  const bf16* sg = X + (size_t)(mbase + srow) * K + schunk * 8;
  const unsigned slds[2] = {
    (unsigned)(uintptr_t)&Abuf[0][srow * AROW + schunk * 8],
    (unsigned)(uintptr_t)&Abuf[1][srow * AROW + schunk * 8] };

  v8f acc[4][2] = {};
  const bf16* wr0 = active ? (W + (size_t)(nbase      + idx) * K + grp * 8) : W;
  const bf16* wr1 = active ? (W + (size_t)(nbase + 16 + idx) * K + grp * 8) : W;

  // prefetch K-chunk 0 into buffer 0
  asm volatile("global_load_async_to_lds_b128 %0, %1, off"
               :: "v"(slds[0]), "v"((unsigned long long)sg) : "memory");

  const int nk = K >> 5;
  for (int ks = 0; ks < nk; ++ks) {
    const int k = ks << 5;
    asm volatile("s_wait_asynccnt 0x0" ::: "memory");   // own staged quarter landed
    __syncthreads();                                    // whole tile published
    if (ks + 1 < nk) {                                  // prefetch next chunk
      asm volatile("global_load_async_to_lds_b128 %0, %1, off"
                   :: "v"(slds[(ks + 1) & 1]),
                      "v"((unsigned long long)(sg + k + 32)) : "memory");
    }
    if (active) {
      FragU b0, b1;
      b0.q[0] = *reinterpret_cast<const I4*>(wr0 + k);
      b0.q[1] = *reinterpret_cast<const I4*>(wr0 + k + 16);
      b1.q[0] = *reinterpret_cast<const I4*>(wr1 + k);
      b1.q[1] = *reinterpret_cast<const I4*>(wr1 + k + 16);
      const bf16* ab = &Abuf[ks & 1][0];
      #pragma unroll
      for (int s = 0; s < 4; ++s) {
        const bf16* ar = ab + (s * 16 + idx) * AROW;
        FragU af;
        af.q[0] = *reinterpret_cast<const I4*>(ar + grp * 8);
        af.q[1] = *reinterpret_cast<const I4*>(ar + 16 + grp * 8);
        acc[s][0] = wmma_bf16(af, b0, acc[s][0]);
        acc[s][1] = wmma_bf16(af, b1, acc[s][1]);
      }
    }
  }

  if (!active) return;
  #pragma unroll
  for (int nt = 0; nt < 2; ++nt) {
    const int col = nbase + nt * 16 + idx;
    const float bv = bias ? bias[col] : 0.f;
    #pragma unroll
    for (int s = 0; s < 4; ++s) {
      #pragma unroll
      for (int v = 0; v < 8; ++v) {
        int row = mbase + s * 16 + grp * 8 + v;   // C/D layout: VGPR v = rows v, v+8
        float r = acc[s][nt][v] + bv;
        if (act == 1) r = (r > 20.f) ? r : log1pf(__expf(r));
        float* yp = Y + (size_t)row * N + col;
        *yp = beta ? (*yp + r) : r;
      }
    }
  }
}

// ---------------------------------------------------------------------------
// Depthwise causal conv (DC=4) + SiLU. rev=1 runs right-to-left (equivalent to
// running the reference on flipped data).
// ---------------------------------------------------------------------------
__global__ void dwconv_silu(const float* __restrict__ xz, const float* __restrict__ Wc,
                            const float* __restrict__ bc, float* __restrict__ u, int rev) {
  size_t i = (size_t)blockIdx.x * blockDim.x + threadIdx.x;   // TOK * DINN
  if (i >= (size_t)TOK * DINN) return;
  int d = (int)(i & (DINN - 1));
  size_t token = i >> 12;
  int l = (int)(token & (LSEQ - 1));
  int b = (int)(token >> 8);
  float acc = bc[d];
  #pragma unroll
  for (int k = 0; k < 4; ++k) {
    int ll = rev ? (l + 3 - k) : (l - 3 + k);
    if (ll >= 0 && ll < LSEQ)
      acc = fmaf(Wc[d * 4 + k], xz[((size_t)(b * LSEQ + ll)) * NPROJ + d], acc);
  }
  u[i] = silu_(acc);
}

// ---------------------------------------------------------------------------
// Selective scan + skip + SiLU(z) gate. One thread per (b,d); B_t/C_t staged in
// LDS once per step (shared by all 4096 d of the same token).
// ---------------------------------------------------------------------------
__global__ __launch_bounds__(256)
void scan_gate(const float* __restrict__ dt, const float* __restrict__ u,
               const float* __restrict__ xdbl, const float* __restrict__ xz,
               const float* __restrict__ Alog, const float* __restrict__ Dp,
               float* __restrict__ y, int rev) {
  int b = blockIdx.x >> 4;                       // 8 * 16 blocks
  int d = ((blockIdx.x & 15) << 8) + threadIdx.x;
  float A[DSTT], s[DSTT];
  #pragma unroll
  for (int j = 0; j < DSTT; ++j) { A[j] = -__expf(Alog[(size_t)d * DSTT + j]); s[j] = 0.f; }
  float Dpv = Dp[d];
  __shared__ float Bs[DSTT], Cs[DSTT];
  for (int t = 0; t < LSEQ; ++t) {
    int tt = rev ? (LSEQ - 1 - t) : t;
    int token = (b << 8) + tt;
    if (threadIdx.x < 32) {
      float v = xdbl[(size_t)token * XPN + DTRR + threadIdx.x];
      if (threadIdx.x < 16) Bs[threadIdx.x] = v; else Cs[threadIdx.x - 16] = v;
    }
    __syncthreads();
    float dtv = dt[(size_t)token * DINN + d];
    float uv  = u[(size_t)token * DINN + d];
    float du  = dtv * uv, accv = 0.f;
    #pragma unroll
    for (int j = 0; j < DSTT; ++j) {
      s[j] = fmaf(s[j], __expf(dtv * A[j]), du * Bs[j]);
      accv = fmaf(s[j], Cs[j], accv);
    }
    float zv = xz[(size_t)token * NPROJ + DINN + d];
    y[(size_t)token * DINN + d] = (accv + uv * Dpv) * silu_(zv);
    __syncthreads();
  }
}

// ---------------------------------------------------------------------------
// Residual add + LayerNorm over DM=2048, one block (256 thr) per token.
// ---------------------------------------------------------------------------
__global__ __launch_bounds__(256)
void add_ln(float* __restrict__ h, const float* __restrict__ o,
            const float* __restrict__ g, const float* __restrict__ bta) {
  __shared__ float red[256];
  size_t base = (size_t)blockIdx.x * DMOD;
  float v[8]; float s = 0.f;
  #pragma unroll
  for (int j = 0; j < 8; ++j) {
    int c = threadIdx.x + j * 256;
    float t = h[base + c] + o[base + c];
    v[j] = t; s += t;
  }
  red[threadIdx.x] = s; __syncthreads();
  for (int st = 128; st > 0; st >>= 1) {
    if (threadIdx.x < st) red[threadIdx.x] += red[threadIdx.x + st];
    __syncthreads();
  }
  float mean = red[0] * (1.f / DMOD);
  __syncthreads();
  float s2 = 0.f;
  #pragma unroll
  for (int j = 0; j < 8; ++j) { float dd = v[j] - mean; s2 = fmaf(dd, dd, s2); }
  red[threadIdx.x] = s2; __syncthreads();
  for (int st = 128; st > 0; st >>= 1) {
    if (threadIdx.x < st) red[threadIdx.x] += red[threadIdx.x + st];
    __syncthreads();
  }
  float rstd = rsqrtf(red[0] * (1.f / DMOD) + 1e-5f);
  #pragma unroll
  for (int j = 0; j < 8; ++j) {
    int c = threadIdx.x + j * 256;
    h[base + c] = (v[j] - mean) * rstd * g[c] + bta[c];
  }
}

// ---------------------------------------------------------------------------
// Decoder 31x31 single-channel conv ('same') over (B, 256, 2048) with LDS tiles.
// ---------------------------------------------------------------------------
__global__ __launch_bounds__(256)
void dec_conv31(const float* __restrict__ h, const float* __restrict__ kw,
                float* __restrict__ out) {
  __shared__ float tile[46][48];
  __shared__ float kws[31 * 31];
  int b  = blockIdx.z;
  int l0 = blockIdx.y * 16, m0 = blockIdx.x * 16;
  int tid = threadIdx.y * 16 + threadIdx.x;
  for (int idx = tid; idx < 46 * 46; idx += 256) {
    int ty = idx / 46, tx = idx % 46;
    int L = l0 - 15 + ty, M = m0 - 15 + tx;
    float v = 0.f;
    if (L >= 0 && L < LSEQ && M >= 0 && M < DMOD)
      v = h[((size_t)(b * LSEQ + L)) * DMOD + M];
    tile[ty][tx] = v;
  }
  for (int idx = tid; idx < 31 * 31; idx += 256) kws[idx] = kw[idx];
  __syncthreads();
  float acc = 0.f;
  for (int p = 0; p < 31; ++p)
    #pragma unroll 31
    for (int q = 0; q < 31; ++q)
      acc = fmaf(tile[threadIdx.y + p][threadIdx.x + q], kws[p * 31 + q], acc);
  out[((size_t)(b * LSEQ + l0 + threadIdx.y)) * DMOD + m0 + threadIdx.x] = acc;
}

// ---------------------------------------------------------------------------
// ConvTranspose (kernel==stride=(2,16) -> non-overlapping): 64-channel dot.
// ---------------------------------------------------------------------------
__global__ void reconstruct(const float* __restrict__ dconv, const float* __restrict__ wt,
                            const float* __restrict__ bt, float* __restrict__ out) {
  size_t i = (size_t)blockIdx.x * blockDim.x + threadIdx.x;   // 8*64*4096
  if (i >= (size_t)BATCH * 64 * 4096) return;
  int c = (int)(i & 4095);
  size_t t = i >> 12;
  int r = (int)(t & 63), b = (int)(t >> 6);
  int g1 = c >> 4, j = c & 15, g0 = r >> 1, ii = r & 1;
  const float* dp = dconv + ((size_t)(b * LSEQ + g1)) * DMOD + g0;
  float acc = bt[0];
  #pragma unroll
  for (int e = 0; e < 64; ++e)
    acc = fmaf(dp[e * 32], wt[e * 32 + ii * 16 + j], acc);
  out[i] = acc;
}

// ---------------------------------------------------------------------------
// Host-side launch
// ---------------------------------------------------------------------------
static inline int blocks_for(size_t n, int bs) { return (int)((n + bs - 1) / bs); }

static inline void launch_gemm(const bf16* X, const bf16* W, const float* bias, float* Y,
                               int M, int N, int K, int beta, int act, hipStream_t stream) {
  dim3 grid(((N >> 5) + 7) / 8, M >> 6);
  gemm_bf16_wmma<<<grid, 256, 0, stream>>>(X, W, bias, Y, M, N, K, beta, act);
}

extern "C" void kernel_launch(void* const* d_in, const int* in_sizes, int n_in,
                              void* d_out, int out_size, void* d_ws, size_t ws_size,
                              hipStream_t stream) {
  const float*         x        = (const float*)d_in[0];
  const unsigned char* mask     = (const unsigned char*)d_in[1];   // jnp bool = 1 byte
  const float*         proj_w   = (const float*)d_in[2];
  const float*         proj_b   = (const float*)d_in[3];
  const float*         pos      = (const float*)d_in[4];
  const float*         in_proj  = (const float*)d_in[5];   // (1,2,8192,2048)
  const float*         conv_w   = (const float*)d_in[6];   // (1,2,4096,4)
  const float*         conv_b   = (const float*)d_in[7];   // (1,2,4096)
  const float*         x_proj   = (const float*)d_in[8];   // (1,2,160,4096)
  const float*         dt_proj  = (const float*)d_in[9];   // (1,2,4096,128)
  const float*         dt_b     = (const float*)d_in[10];  // (1,2,4096)
  const float*         A_log    = (const float*)d_in[11];  // (1,2,4096,16)
  const float*         Dp       = (const float*)d_in[12];  // (1,2,4096)
  const float*         out_proj = (const float*)d_in[13];  // (1,2,2048,4096)
  const float*         ln_g     = (const float*)d_in[14];
  const float*         ln_b     = (const float*)d_in[15];
  const float*         dec_w    = (const float*)d_in[16];  // (1,1,31,31)
  const float*         convT_w  = (const float*)d_in[17];  // (64,1,2,16)
  const float*         convT_b  = (const float*)d_in[18];  // (1,)
  float* outp = (float*)d_out;

  // workspace arena (256 B aligned slots)
  char* p = (char*)d_ws;
  auto alloc = [&](size_t bytes) { char* r = p; p += (bytes + 255) & ~(size_t)255; return r; };
  float* h     = (float*)alloc((size_t)TOK * DMOD * 4);
  float* osum  = (float*)alloc((size_t)TOK * DMOD * 4);
  float* xz    = (float*)alloc((size_t)TOK * NPROJ * 4);
  float* u     = (float*)alloc((size_t)TOK * DINN * 4);
  float* dt    = (float*)alloc((size_t)TOK * DINN * 4);
  float* y     = (float*)alloc((size_t)TOK * DINN * 4);
  float* dconv = (float*)alloc((size_t)TOK * DMOD * 4);
  float* xdbl  = (float*)alloc((size_t)TOK * XPN * 4);
  bf16*  h_bf  = (bf16*)alloc((size_t)TOK * DMOD * 2);
  bf16*  u_bf  = (bf16*)alloc((size_t)TOK * DINN * 2);
  bf16*  y_bf  = (bf16*)alloc((size_t)TOK * DINN * 2);
  bf16*  dt_bf = (bf16*)alloc((size_t)TOK * DTRR * 2);
  bf16*  wbuf  = (bf16*)alloc((size_t)NPROJ * DMOD * 2);   // reused: Win then Wout
  bf16*  wx_bf = (bf16*)alloc((size_t)XPN * DINN * 2);
  bf16*  wdt_bf= (bf16*)alloc((size_t)DINN * DTRR * 2);

  const int BS = 256;

  // 1) patch embed -> h (TOK x DM), then h in bf16 (shared by both directions)
  patch_embed<<<blocks_for((size_t)TOK * DMOD, BS), BS, 0, stream>>>(x, mask, proj_w, proj_b, pos, h);
  cvt_bf16<<<blocks_for((size_t)TOK * DMOD, BS), BS, 0, stream>>>(h, h_bf, (size_t)TOK * DMOD);

  for (int dir = 0; dir < 2; ++dir) {
    const float* Win  = in_proj  + (size_t)dir * NPROJ * DMOD;
    const float* Wc   = conv_w   + (size_t)dir * DINN * 4;
    const float* bc   = conv_b   + (size_t)dir * DINN;
    const float* Wx   = x_proj   + (size_t)dir * XPN * DINN;
    const float* Wdt  = dt_proj  + (size_t)dir * DINN * DTRR;
    const float* bdt  = dt_b     + (size_t)dir * DINN;
    const float* Al   = A_log    + (size_t)dir * DINN * DSTT;
    const float* Dpd  = Dp       + (size_t)dir * DINN;
    const float* Wout = out_proj + (size_t)dir * DMOD * DINN;

    // in_proj: xz = h * Win^T   (M=2048, N=8192, K=2048)
    cvt_bf16<<<blocks_for((size_t)NPROJ * DMOD, BS), BS, 0, stream>>>(Win, wbuf, (size_t)NPROJ * DMOD);
    launch_gemm(h_bf, wbuf, nullptr, xz, TOK, NPROJ, DMOD, 0, 0, stream);

    // depthwise conv + silu -> u
    dwconv_silu<<<blocks_for((size_t)TOK * DINN, BS), BS, 0, stream>>>(xz, Wc, bc, u, dir);

    // x_proj: xdbl = u * Wx^T   (M=2048, N=160, K=4096)
    cvt_bf16<<<blocks_for((size_t)TOK * DINN, BS), BS, 0, stream>>>(u, u_bf, (size_t)TOK * DINN);
    cvt_bf16<<<blocks_for((size_t)XPN * DINN, BS), BS, 0, stream>>>(Wx, wx_bf, (size_t)XPN * DINN);
    launch_gemm(u_bf, wx_bf, nullptr, xdbl, TOK, XPN, DINN, 0, 0, stream);

    // dt = softplus(xdbl[:, :128] * Wdt^T + bdt)   (M=2048, N=4096, K=128)
    cvt_bf16_strided<<<blocks_for((size_t)TOK * DTRR, BS), BS, 0, stream>>>(xdbl, dt_bf, TOK, DTRR, XPN);
    cvt_bf16<<<blocks_for((size_t)DINN * DTRR, BS), BS, 0, stream>>>(Wdt, wdt_bf, (size_t)DINN * DTRR);
    launch_gemm(dt_bf, wdt_bf, bdt, dt, TOK, DINN, DTRR, 0, 1, stream);

    // selective scan + skip + gate -> y
    scan_gate<<<BATCH * (DINN / 256), BS, 0, stream>>>(dt, u, xdbl, xz, Al, Dpd, y, dir);

    // out_proj: osum (+)= y * Wout^T   (M=2048, N=2048, K=4096)
    cvt_bf16<<<blocks_for((size_t)TOK * DINN, BS), BS, 0, stream>>>(y, y_bf, (size_t)TOK * DINN);
    cvt_bf16<<<blocks_for((size_t)DMOD * DINN, BS), BS, 0, stream>>>(Wout, wbuf, (size_t)DMOD * DINN);
    launch_gemm(y_bf, wbuf, nullptr, osum, TOK, DMOD, DINN, dir, 0, stream);
  }

  // residual + layernorm (in place on h)
  add_ln<<<TOK, BS, 0, stream>>>(h, osum, ln_g, ln_b);

  // decoder 31x31 conv
  {
    dim3 grid(DMOD / 16, LSEQ / 16, BATCH), blk(16, 16);
    dec_conv31<<<grid, blk, 0, stream>>>(h, dec_w, dconv);
  }

  // transposed conv -> rec (first output), then copy x (second output)
  reconstruct<<<blocks_for((size_t)BATCH * 64 * 4096, BS), BS, 0, stream>>>(dconv, convT_w, convT_b, outp);
  copy_f32<<<blocks_for((size_t)BATCH * 64 * 4096, BS), BS, 0, stream>>>(x, outp + (size_t)BATCH * 64 * 4096,
                                                                         (size_t)BATCH * 64 * 4096);
}
